// _VarPDE_43181601194893
// MI455X (gfx1250) — matrix operand order repaired and checked by reading, not compiled, not run on hardware
//
#include <hip/hip_runtime.h>
#include <hip/hip_bf16.h>
#include <stdint.h>

// Problem constants (match reference)
#define B_     8
#define N_     1024
#define D_     256
#define L_     512
#define K_TOP  16
#define NSTEPS 8

typedef __attribute__((ext_vector_type(16))) __bf16 v16bf;
typedef __attribute__((ext_vector_type(8)))  float  v8f;
typedef __attribute__((ext_vector_type(4)))  int    v4i;

// ---------------- async copy to LDS (gfx1250) with sync fallback ------------
#if defined(__has_builtin)
#  if __has_builtin(__builtin_amdgcn_global_load_async_to_lds_b128)
#    define HAVE_ASYNC_CP 1
#  endif
#endif
#ifndef HAVE_ASYNC_CP
#  define HAVE_ASYNC_CP 0
#endif

typedef __attribute__((address_space(1))) v4i g_v4i;   // global int4
typedef __attribute__((address_space(3))) v4i l_v4i;   // LDS int4

__device__ __forceinline__ void cp16(void* lds_dst, const void* g_src) {
#if HAVE_ASYNC_CP
  __builtin_amdgcn_global_load_async_to_lds_b128((g_v4i*)g_src, (l_v4i*)lds_dst, 0, 0);
#else
  *(uint4*)lds_dst = *(const uint4*)g_src;
#endif
}
__device__ __forceinline__ void cp_wait() {
#if HAVE_ASYNC_CP
#  if __has_builtin(__builtin_amdgcn_s_wait_asynccnt)
  __builtin_amdgcn_s_wait_asynccnt(0);
#  else
  asm volatile("s_wait_asynccnt 0" ::: "memory");
#  endif
#endif
}

// ---------------- helpers ----------------
__device__ __forceinline__ unsigned short f2bf(float f) {
  unsigned u = __builtin_bit_cast(unsigned, f);
  unsigned r = u + 0x7FFFu + ((u >> 16) & 1u);   // round-to-nearest-even
  return (unsigned short)(r >> 16);
}

struct U2 { uint4 a, b; };  // 32 bytes == v16bf

// A fragment (16x32 bf16, ISA 7.12.2): lane m = lane&15, half = lane>>4,
// elements K = {8h..8h+7} in VGPR0-3, {16+8h..23+8h} in VGPR4-7 -> two b128 LDS reads.
__device__ __forceinline__ v16bf load_frag_A(const unsigned short* lds, int ldl,
                                             int mbase, int lane) {
  int half = (lane >> 4) & 1;
  const unsigned short* row = lds + (mbase + (lane & 15)) * ldl;
  U2 u;
  u.a = *(const uint4*)(row + 8 * half);
  u.b = *(const uint4*)(row + 16 + 8 * half);
  return __builtin_bit_cast(v16bf, u);
}

// B fragment (32x16 bf16): lane n = lane&15, K = {8h..8h+7} then {16+8h..23+8h}
// at row n of the K-major LDS tile -> two b128 LDS reads (same order as A).
__device__ __forceinline__ v16bf load_frag_B(const unsigned short* lds, int ldl,
                                             int nbase, int lane) {
  int half = (lane >> 4) & 1;
  const unsigned short* row = lds + (nbase + (lane & 15)) * ldl;
  U2 u;
  u.a = *(const uint4*)(row + 8 * half);
  u.b = *(const uint4*)(row + 16 + 8 * half);
  return __builtin_bit_cast(v16bf, u);
}

// ---------------- shared GEMM core ------------------------------------------
// C[128x128] block tile, K chunks of 32, 8 waves: wave grid 4(M) x 2(N),
// wave tile 32x64 (2x4 wmma 16x16 tiles). A is [M,K] row-major bf16;
// B is given TRANSPOSED: Bt[Ncol][K] row-major bf16 -> all staging is
// plain 16-byte copies (async to LDS, double buffered).
__device__ __forceinline__ void gemm_block(const unsigned short* __restrict__ Ag, int lda,
                                           const unsigned short* __restrict__ Bt, int ldbt,
                                           int Kdim, int mblk, int nblk,
                                           v8f (&acc)[2][4]) {
  __shared__ __align__(16) unsigned short lA[2][128][48];   // K stride 48 (96B rows)
  __shared__ __align__(16) unsigned short lB[2][128][48];
  const int tid  = threadIdx.x;
  const int lane = tid & 31;
  const int w    = tid >> 5;
  const int wm   = w >> 1;                   // 0..3
  const int wn   = w & 1;                    // 0..1
  // staging coords at uint4 granularity (512 vectors per tile, 2 per thread)
  const int r0 = tid >> 2,           c0 = (tid & 3) * 8;
  const int r1 = (tid + 256) >> 2,   c1 = ((tid + 256) & 3) * 8;

  auto stage = [&](int buf, int k0) {
    cp16(&lA[buf][r0][c0], Ag + (size_t)(mblk + r0) * lda  + (k0 + c0));
    cp16(&lA[buf][r1][c1], Ag + (size_t)(mblk + r1) * lda  + (k0 + c1));
    cp16(&lB[buf][r0][c0], Bt + (size_t)(nblk + r0) * ldbt + (k0 + c0));
    cp16(&lB[buf][r1][c1], Bt + (size_t)(nblk + r1) * ldbt + (k0 + c1));
  };

  stage(0, 0);
  cp_wait();
  __syncthreads();
  int buf = 0;
  for (int k0 = 0; k0 < Kdim; k0 += 32) {
    if (k0 + 32 < Kdim) stage(buf ^ 1, k0 + 32);   // overlap copy(t+1) with compute(t)

    v16bf a0 = load_frag_A(&lA[buf][0][0], 48, wm * 32 + 0,  lane);
    v16bf a1 = load_frag_A(&lA[buf][0][0], 48, wm * 32 + 16, lane);
#pragma unroll
    for (int ni = 0; ni < 4; ++ni) {
      v16bf bfr = load_frag_B(&lB[buf][0][0], 48, wn * 64 + ni * 16, lane);
      acc[0][ni] = __builtin_amdgcn_wmma_f32_16x16x32_bf16(
          false, a0, false, bfr, (short)0, acc[0][ni], false, false);
      acc[1][ni] = __builtin_amdgcn_wmma_f32_16x16x32_bf16(
          false, a1, false, bfr, (short)0, acc[1][ni], false, false);
    }
    cp_wait();
    __syncthreads();
    buf ^= 1;
  }
}

#define ZERO_ACC(acc)                                   \
  _Pragma("unroll") for (int _i = 0; _i < 2; ++_i)      \
  _Pragma("unroll") for (int _j = 0; _j < 4; ++_j)      \
  _Pragma("unroll") for (int _e = 0; _e < 8; ++_e) acc[_i][_j][_e] = 0.0f;

// ---------- kernel 1: per-row normalize, fold 1/sqrt(L-1), emit bf16 --------
__global__ __launch_bounds__(256) void norm_rows(const float* __restrict__ x,
                                                 unsigned short* __restrict__ xn) {
  int row = blockIdx.x;                      // B*N rows
  const float* src = x + (size_t)row * L_;
  int tid = threadIdx.x;
  float v0 = src[tid], v1 = src[tid + 256];
  float s = v0 + v1, ss = v0 * v0 + v1 * v1;
#pragma unroll
  for (int o = 16; o > 0; o >>= 1) { s += __shfl_xor(s, o); ss += __shfl_xor(ss, o); }
  __shared__ float sh[18];
  int wv = tid >> 5, lane = tid & 31;
  if (lane == 0) { sh[wv] = s; sh[wv + 8] = ss; }
  __syncthreads();
  if (tid == 0) {
    float S = 0.f, SS = 0.f;
#pragma unroll
    for (int i = 0; i < 8; ++i) { S += sh[i]; SS += sh[i + 8]; }
    sh[16] = S; sh[17] = SS;
  }
  __syncthreads();
  float mean = sh[16] * (1.0f / L_);
  float var  = fmaxf((sh[17] - (float)L_ * mean * mean) / (float)(L_ - 1), 0.0f);
  float scale = (1.0f / (sqrtf(var) + 1e-6f)) * rsqrtf((float)(L_ - 1));
  unsigned short* dst = xn + (size_t)row * L_;
  dst[tid]       = f2bf((v0 - mean) * scale);
  dst[tid + 256] = f2bf((v1 - mean) * scale);
}

// ---------- kernel 2: corr = xn xn^T (Bt == xn, already [node][L]) ----------
__global__ __launch_bounds__(256) void corr_gemm(const unsigned short* __restrict__ xn,
                                                 float* __restrict__ corr) {
  int b = blockIdx.z;
  int mblk = blockIdx.y * 128, nblk = blockIdx.x * 128;
  const unsigned short* Ag = xn + (size_t)b * N_ * L_;
  v8f acc[2][4];
  ZERO_ACC(acc)
  gemm_block(Ag, L_, Ag, L_, L_, mblk, nblk, acc);

  int tid = threadIdx.x, lane = tid & 31, w = tid >> 5, wm = w >> 1, wn = w & 1;
  int half = lane >> 4, nloc = lane & 15;
  float* Cb = corr + (size_t)b * N_ * N_;
#pragma unroll
  for (int mi = 0; mi < 2; ++mi)
#pragma unroll
    for (int ni = 0; ni < 4; ++ni)
#pragma unroll
      for (int r = 0; r < 8; ++r) {
        int m = mblk + wm * 32 + mi * 16 + r + 8 * half;
        int n = nblk + wn * 64 + ni * 16 + nloc;
        Cb[(size_t)m * N_ + n] = acc[mi][ni][r];
      }
}

// ---------- kernel 3: per-row top-(K+1) mask, in place (one wave32 per row) -
__global__ __launch_bounds__(256) void topk_mask(float* __restrict__ corr) {
  int wave = threadIdx.x >> 5, lane = threadIdx.x & 31;
  int row = blockIdx.x * 8 + wave;           // 0..B*N-1
  float* p = corr + (size_t)row * N_;
  float v[32];
#pragma unroll
  for (int j = 0; j < 32; ++j) v[j] = p[lane + 32 * j];
  unsigned kept = 0u;
  for (int it = 0; it < K_TOP + 1; ++it) {
    float lm = -3.4e38f; int la = -1;
#pragma unroll
    for (int j = 0; j < 32; ++j) {
      bool avail = ((kept >> j) & 1u) == 0u;
      if (avail && v[j] > lm) { lm = v[j]; la = j; }
    }
    float wmx = lm;
#pragma unroll
    for (int o = 16; o > 0; o >>= 1) wmx = fmaxf(wmx, __shfl_xor(wmx, o));
    unsigned long long bm = __ballot((lm == wmx) && (la >= 0));
    int src = (int)(__ffsll(bm) - 1);
    if (lane == src) kept |= (1u << la);
  }
#pragma unroll
  for (int j = 0; j < 32; ++j)
    p[lane + 32 * j] = ((kept >> j) & 1u) ? v[j] : 0.0f;
}

// ---------- kernel 4: deg of 0.5(M+M^T)+I ; dinv = deg^-1/2 -----------------
__global__ __launch_bounds__(256) void degree_dinv(const float* __restrict__ M,
                                                   float* __restrict__ dinv) {
  int wave = threadIdx.x >> 5, lane = threadIdx.x & 31;
  int row = blockIdx.x * 8 + wave;
  int b = row >> 10, i = row & (N_ - 1);
  const float* Mb = M + (size_t)b * N_ * N_;
  float rs = 0.f, cs = 0.f;
  for (int j = lane; j < N_; j += 32) {
    rs += Mb[(size_t)i * N_ + j];
    cs += Mb[(size_t)j * N_ + i];
  }
  float s = 0.5f * (rs + cs);
#pragma unroll
  for (int o = 16; o > 0; o >>= 1) s += __shfl_xor(s, o);
  if (lane == 0) dinv[row] = rsqrtf(fmaxf(s + 1.0f, 1e-6f));
}

// ---------- kernel 5: Ahat = Dinv (0.5(M+M^T)+I) Dinv -> bf16 ---------------
__global__ __launch_bounds__(256) void build_ahat(const float* __restrict__ M,
                                                  const float* __restrict__ dinv,
                                                  unsigned short* __restrict__ ahat) {
  size_t idx = (size_t)blockIdx.x * 256 + threadIdx.x;  // over B*N*N
  int b = (int)(idx >> 20);
  int rem = (int)(idx & (N_ * N_ - 1));
  int i = rem >> 10, j = rem & (N_ - 1);
  const float* Mb = M + (size_t)b * N_ * N_;
  float v = 0.5f * (Mb[rem] + Mb[(size_t)j * N_ + i]) + (i == j ? 1.0f : 0.0f);
  ahat[idx] = f2bf(v * dinv[b * N_ + i] * dinv[b * N_ + j]);
}

// ---------- kernel 6: state init; xtbT is [B][D][N] bf16 --------------------
__global__ __launch_bounds__(256) void init_state(const float* __restrict__ tokens,
                                                  float* __restrict__ x,
                                                  float* __restrict__ xt,
                                                  unsigned short* __restrict__ xtbT) {
  size_t idx = (size_t)blockIdx.x * 256 + threadIdx.x;  // b*N*D + n*D + d
  float t = tokens[idx];
  x[idx] = t; xt[idx] = t;
  int b = (int)(idx >> 18);
  int rem = (int)(idx & (N_ * D_ - 1));
  int n = rem >> 8, d = rem & (D_ - 1);
  xtbT[((size_t)b * D_ + d) * N_ + n] = f2bf(t);
}

__global__ __launch_bounds__(256) void build_wb(const float* __restrict__ W,
                                                unsigned short* __restrict__ WbT) {
  int idx = blockIdx.x * 256 + threadIdx.x;   // WbT[dout][din] = W[dout][din]
  WbT[idx] = f2bf(W[idx]);
}

// ---------- kernel 7: y = Ahat @ xt (per batch); Bt = xtbT[b] ---------------
__global__ __launch_bounds__(256) void gemm_ax(const unsigned short* __restrict__ ahat,
                                               const unsigned short* __restrict__ xtbT,
                                               float* __restrict__ y,
                                               unsigned short* __restrict__ yb) {
  int b = blockIdx.z;
  int mblk = blockIdx.y * 128, nblk = blockIdx.x * 128;
  const unsigned short* Ag = ahat + (size_t)b * N_ * N_;
  const unsigned short* Bt = xtbT + (size_t)b * D_ * N_;   // [d][node]
  v8f acc[2][4];
  ZERO_ACC(acc)
  gemm_block(Ag, N_, Bt, N_, N_, mblk, nblk, acc);

  int tid = threadIdx.x, lane = tid & 31, w = tid >> 5, wm = w >> 1, wn = w & 1;
  int half = lane >> 4, nloc = lane & 15;
  float* Yb = y + (size_t)b * N_ * D_;
  unsigned short* Ybb = yb + (size_t)b * N_ * D_;
#pragma unroll
  for (int mi = 0; mi < 2; ++mi)
#pragma unroll
    for (int ni = 0; ni < 4; ++ni)
#pragma unroll
      for (int r = 0; r < 8; ++r) {
        int m = mblk + wm * 32 + mi * 16 + r + 8 * half;
        int n = nblk + wn * 64 + ni * 16 + nloc;
        float vv = acc[mi][ni][r];
        Yb[(size_t)m * D_ + n] = vv;
        Ybb[(size_t)m * D_ + n] = f2bf(vv);
      }
}

// ---------- kernel 8: z = y @ W^T with fused RK4-stage epilogue -------------
// k = alpha*(xt - y) + z ;  stage<3: acc,xt_next ; stage==3: x update
__global__ __launch_bounds__(256) void gemm_react_rk4(
    const unsigned short* __restrict__ yb, const unsigned short* __restrict__ WbT,
    const float* __restrict__ y, float* __restrict__ x, float* __restrict__ accb,
    float* __restrict__ xt, unsigned short* __restrict__ xtbT,
    const float* __restrict__ alpha_raw, int stage, float dt) {
  int mblk = blockIdx.y * 128, nblk = blockIdx.x * 128;   // M = B*N flat, Ncol = D
  v8f acc[2][4];
  ZERO_ACC(acc)
  gemm_block(yb, D_, WbT, D_, D_, mblk, nblk, acc);

  float alpha = fminf(log1pf(__expf(alpha_raw[0])), 2.0f);  // softplus, clamped
  int tid = threadIdx.x, lane = tid & 31, w = tid >> 5, wm = w >> 1, wn = w & 1;
  int half = lane >> 4, nloc = lane & 15;
#pragma unroll
  for (int mi = 0; mi < 2; ++mi)
#pragma unroll
    for (int ni = 0; ni < 4; ++ni)
#pragma unroll
      for (int r = 0; r < 8; ++r) {
        int m = mblk + wm * 32 + mi * 16 + r + 8 * half;
        int n = nblk + wn * 64 + ni * 16 + nloc;
        size_t off = (size_t)m * D_ + n;
        int b = m >> 10, node = m & (N_ - 1);
        size_t offT = ((size_t)b * D_ + n) * N_ + node;
        float z = acc[mi][ni][r];
        float k = alpha * (xt[off] - y[off]) + z;
        float xn2;
        if (stage == 0)      { accb[off] = k;          xn2 = x[off] + 0.5f * dt * k; }
        else if (stage == 1) { accb[off] += 2.0f * k;  xn2 = x[off] + 0.5f * dt * k; }
        else if (stage == 2) { accb[off] += 2.0f * k;  xn2 = x[off] + dt * k; }
        else                 { xn2 = x[off] + (dt / 6.0f) * (accb[off] + k); x[off] = xn2; }
        xt[off] = xn2;
        xtbT[offT] = f2bf(xn2);
      }
}

// ---------- kernel 9: out = relu(x) -----------------------------------------
__global__ __launch_bounds__(256) void relu_out(const float* __restrict__ x,
                                                float* __restrict__ out) {
  size_t idx = (size_t)blockIdx.x * 256 + threadIdx.x;
  out[idx] = fmaxf(x[idx], 0.0f);
}

// ---------- host launch ------------------------------------------------------
extern "C" void kernel_launch(void* const* d_in, const int* in_sizes, int n_in,
                              void* d_out, int out_size, void* d_ws, size_t ws_size,
                              hipStream_t stream) {
  const float* tokens    = (const float*)d_in[0];  // [B,N,D]
  const float* x_bn      = (const float*)d_in[1];  // [B,N,L]
  const float* theta_W   = (const float*)d_in[2];  // [D,D]
  const float* alpha_raw = (const float*)d_in[3];  // [1]

  char* ws = (char*)d_ws;
  size_t off = 0;
  auto alloc = [&](size_t bytes) -> void* {
    void* p = ws + off;
    off = (off + bytes + 255) & ~(size_t)255;
    return p;
  };
  unsigned short* xn    = (unsigned short*)alloc((size_t)B_ * N_ * L_ * 2);
  float*          corr  = (float*)         alloc((size_t)B_ * N_ * N_ * 4);
  unsigned short* ahat  = (unsigned short*)alloc((size_t)B_ * N_ * N_ * 2);
  float*          dinv  = (float*)         alloc((size_t)B_ * N_ * 4);
  unsigned short* WbT   = (unsigned short*)alloc((size_t)D_ * D_ * 2);
  float*          xbuf  = (float*)         alloc((size_t)B_ * N_ * D_ * 4);
  float*          xt    = (float*)         alloc((size_t)B_ * N_ * D_ * 4);
  unsigned short* xtbT  = (unsigned short*)alloc((size_t)B_ * N_ * D_ * 2);
  float*          y     = (float*)         alloc((size_t)B_ * N_ * D_ * 4);
  unsigned short* yb    = (unsigned short*)alloc((size_t)B_ * N_ * D_ * 2);
  float*          accb  = (float*)         alloc((size_t)B_ * N_ * D_ * 4);

  const float dt = (0.2f - 0.0f) / (float)NSTEPS;

  norm_rows<<<B_ * N_, 256, 0, stream>>>(x_bn, xn);
  corr_gemm<<<dim3(N_ / 128, N_ / 128, B_), 256, 0, stream>>>(xn, corr);
  topk_mask<<<(B_ * N_) / 8, 256, 0, stream>>>(corr);
  degree_dinv<<<(B_ * N_) / 8, 256, 0, stream>>>(corr, dinv);
  build_ahat<<<(B_ * N_ * N_) / 256, 256, 0, stream>>>(corr, dinv, ahat);
  init_state<<<(B_ * N_ * D_) / 256, 256, 0, stream>>>(tokens, xbuf, xt, xtbT);
  build_wb<<<(D_ * D_) / 256, 256, 0, stream>>>(theta_W, WbT);

  for (int step = 0; step < NSTEPS; ++step) {
    for (int stage = 0; stage < 4; ++stage) {
      gemm_ax<<<dim3(D_ / 128, N_ / 128, B_), 256, 0, stream>>>(ahat, xtbT, y, yb);
      gemm_react_rk4<<<dim3(D_ / 128, (B_ * N_) / 128, 1), 256, 0, stream>>>(
          yb, WbT, y, xbuf, accb, xt, xtbT, alpha_raw, stage, dt);
    }
  }
  relu_out<<<(B_ * N_ * D_) / 256, 256, 0, stream>>>(xbuf, (float*)d_out);
}
